// FAGAT_16527034155009
// MI455X (gfx1250) — compile-verified, matrix operand order checked by reading, not run on hardware
//
#include <hip/hip_runtime.h>
#include <cstdint>
#include <cstddef>

#define NEG_SLOPE 0.2f
#define EPSV 1e-16f

#define NN 50000
#define EE 300000
#define ETOT (EE + NN)
#define H1 4
#define H2 2
#define DD 64
#define C1 (H1 * DD)   // 256
#define C2 (H2 * DD)   // 128
#define K1PAD 32       // 27 padded to 32
#define KMAX 256
#define LDS_STRIDE (KMAX + 8)   // halves; +8 staggers banks (528B row pitch)

typedef __attribute__((ext_vector_type(16))) _Float16 v16h;
typedef __attribute__((ext_vector_type(8)))  _Float16 v8h;
typedef __attribute__((ext_vector_type(8)))  float    v8f;

// ---------------------------------------------------------------- utilities
__device__ __forceinline__ void atomicMaxF(float* addr, float val) {
  int* ia = (int*)addr;
  int cur = *ia;
  while (__int_as_float(cur) < val) {
    int prev = atomicCAS(ia, cur, __float_as_int(val));
    if (prev == cur) break;
    cur = prev;
  }
}

__global__ void k_fill(float* __restrict__ p, float v, long n) {
  long i = (long)blockIdx.x * blockDim.x + threadIdx.x;
  if (i < n) p[i] = v;
}

// x [N x Kin] f32 -> x16 [N x Kpad] f16 (zero padded)
__global__ void k_convert_x_pad(const float* __restrict__ x, _Float16* __restrict__ x16,
                                int n, int Kin, int Kpad) {
  long i = (long)blockIdx.x * blockDim.x + threadIdx.x;
  long tot = (long)n * Kpad;
  if (i >= tot) return;
  int row = (int)(i / Kpad), k = (int)(i % Kpad);
  x16[i] = (k < Kin) ? (_Float16)x[(long)row * Kin + k] : (_Float16)0.0f;
}

// W [Kin x Ncol] f32 -> Wt16 [Ncol x Kpad] f16 transposed, zero padded
__global__ void k_convert_wt(const float* __restrict__ W, _Float16* __restrict__ Wt,
                             int Kin, int Ncol, int Kpad) {
  long i = (long)blockIdx.x * blockDim.x + threadIdx.x;
  long tot = (long)Ncol * Kpad;
  if (i >= tot) return;
  int col = (int)(i / Kpad), k = (int)(i % Kpad);
  Wt[i] = (k < Kin) ? (_Float16)W[(long)k * Ncol + col] : (_Float16)0.0f;
}

// ---------------------------------------------------------------- WMMA GEMM
// C[M x Ncol] = A[M x K](f16 row major) * Bt[Ncol x K](f16, B col-major).
// Block: 256 threads = 8 waves -> one 16-row x 128-col tile; A tile staged in LDS
// once per block (waves share it), B fragments stream from global (L2-resident).
// Requires: M%16==0, K%32==0, K<=KMAX, Ncol == 128*gridDim.y.
__global__ void k_gemm_wmma(const _Float16* __restrict__ A,
                            const _Float16* __restrict__ Bt,
                            float* __restrict__ C,
                            int M, int K, int Ncol) {
  __shared__ _Float16 As[16 * LDS_STRIDE];

  const int tid  = threadIdx.x;
  const int lane = tid & 31;
  const int wave = tid >> 5;
  const int row0 = blockIdx.x << 4;
  const int col0 = ((blockIdx.y << 3) + wave) << 4;

  // cooperative load of A tile [16 x K] into LDS, 8 halves (16B) per step
  const int tileElems = 16 * K;
  for (int idx = tid * 8; idx < tileElems; idx += 256 * 8) {
    int r = idx / K, c = idx % K;
    *(v8h*)(As + r * LDS_STRIDE + c) = *(const v8h*)(A + (size_t)(row0 + r) * K + c);
  }
  __syncthreads();

  const int m    = lane & 15;
  const int half = lane >> 4;   // 0: K-halves {0..7,16..23}; 1: {8..15,24..31}
  const _Float16* arow = As + m * LDS_STRIDE;
  const _Float16* bcol = Bt + (size_t)(col0 + m) * K;

  v8f acc = {};
  for (int k0 = 0; k0 < K; k0 += 32) {
    v8h alo = *(const v8h*)(arow + k0 + half * 8);        // ds_load_b128
    v8h ahi = *(const v8h*)(arow + k0 + 16 + half * 8);   // ds_load_b128
    v8h blo = *(const v8h*)(bcol + k0 + half * 8);
    v8h bhi = *(const v8h*)(bcol + k0 + 16 + half * 8);
    v16h a, b;
#pragma unroll
    for (int i = 0; i < 8; ++i) {
      a[i] = alo[i]; a[i + 8] = ahi[i];
      b[i] = blo[i]; b[i + 8] = bhi[i];
    }
    acc = __builtin_amdgcn_wmma_f32_16x16x32_f16(
        /*neg_a=*/false, a, /*neg_b=*/false, b,
        /*c_mod=*/(short)0, acc, /*reuse_a=*/false, /*reuse_b=*/false);
  }
  // C/D layout: lanes 0-15 -> col=lane, rows row0+r; lanes 16-31 -> rows row0+8+r
  const int ccol = col0 + m;
#pragma unroll
  for (int r = 0; r < 8; ++r)
    C[(size_t)(row0 + half * 8 + r) * Ncol + ccol] = acc[r];
}

// ---------------------------------------------------------------- GAT pieces
// per-node attention half scores: s[n,h] = dot(h[n,h*D:...], a[h,:])
__global__ void k_scores(const float* __restrict__ hfeat,
                         const float* __restrict__ a_src, const float* __restrict__ a_dst,
                         float* __restrict__ s_src, float* __restrict__ s_dst,
                         int n, int H) {
  long i = (long)blockIdx.x * blockDim.x + threadIdx.x;
  if (i >= (long)n * H) return;
  int node = (int)(i / H), h = (int)(i % H);
  const float4* hp = (const float4*)(hfeat + (size_t)node * H * DD + h * DD);
  const float4* as = (const float4*)(a_src + h * DD);
  const float4* ad = (const float4*)(a_dst + h * DD);
  float ss = 0.f, sd = 0.f;
#pragma unroll
  for (int k = 0; k < DD / 4; ++k) {
    float4 v = hp[k], a = as[k], d = ad[k];
    ss += v.x * a.x + v.y * a.y + v.z * a.z + v.w * a.w;
    sd += v.x * d.x + v.y * d.y + v.z * d.z + v.w * d.w;
  }
  s_src[i] = ss;
  s_dst[i] = sd;
}

// pass 1: per-edge leaky-relu logits + segment max over dst
__global__ void k_edge_max(const int* __restrict__ src, const int* __restrict__ dst,
                           const float* __restrict__ s_src, const float* __restrict__ s_dst,
                           float* __restrict__ logits, float* __restrict__ mbuf, int H) {
  long i = (long)blockIdx.x * blockDim.x + threadIdx.x;
  long tot = (long)ETOT * H;
  if (i >= tot) return;
  int e = (int)(i / H), h = (int)(i % H);
  int s, d;
  if (e < EE) { s = src[e]; d = dst[e]; } else { s = d = e - EE; }
  float x = s_src[(long)s * H + h] + s_dst[(long)d * H + h];
  float lg = x > 0.f ? x : x * NEG_SLOPE;
  logits[i] = lg;
  atomicMaxF(&mbuf[(long)d * H + h], lg);
}

// pass 2: e = exp(logit - m[dst]); z[dst] += e  (store e in place of logit)
__global__ void k_edge_expsum(const int* __restrict__ dst,
                              float* __restrict__ logits,
                              const float* __restrict__ mbuf, float* __restrict__ zbuf, int H) {
  long i = (long)blockIdx.x * blockDim.x + threadIdx.x;
  long tot = (long)ETOT * H;
  if (i >= tot) return;
  int e = (int)(i / H), h = (int)(i % H);
  int d = (e < EE) ? dst[e] : (e - EE);
  float ev = __expf(logits[i] - mbuf[(long)d * H + h]);
  logits[i] = ev;
  atomicAdd(&zbuf[(long)d * H + h], ev);
}

// pass 3: out[dst,f] += h[src,f] * e/(z[dst]+eps). one block per edge, thread per feature.
__global__ void k_edge_aggregate(const int* __restrict__ src, const int* __restrict__ dst,
                                 const float* __restrict__ expv, const float* __restrict__ zbuf,
                                 const float* __restrict__ hfeat, float* __restrict__ out, int H) {
  int e = blockIdx.x;
  int f = threadIdx.x;                 // blockDim.x == H*DD
  int s, d;
  if (e < EE) { s = src[e]; d = dst[e]; } else { s = d = e - EE; }
  int h = f >> 6;                      // f / DD
  float ev = expv[(long)e * H + h];
  float alpha = ev / (zbuf[(long)d * H + h] + EPSV);
  int C = H * DD;
  atomicAdd(&out[(size_t)d * C + f], hfeat[(size_t)s * C + f] * alpha);
}

// h' = elu(out + bias); also emit fp16 copy for next WMMA GEMM
__global__ void k_elu_bias_f16(const float* __restrict__ in, const float* __restrict__ bias,
                               _Float16* __restrict__ out16, int n, int C) {
  long i = (long)blockIdx.x * blockDim.x + threadIdx.x;
  long tot = (long)n * C;
  if (i >= tot) return;
  int c = (int)(i % C);
  float v = in[i] + bias[c];
  v = v > 0.f ? v : (__expf(v) - 1.0f);
  out16[i] = (_Float16)v;
}

// y[n] = sigmoid( sum_f elu(out2[n,f]+b2[f]) * Wfc[f] + bfc )
__global__ void k_final(const float* __restrict__ out2, const float* __restrict__ b2,
                        const float* __restrict__ Wfc, const float* __restrict__ bfc,
                        float* __restrict__ y, int n) {
  long i = (long)blockIdx.x * blockDim.x + threadIdx.x;
  if (i >= n) return;
  const float4* row = (const float4*)(out2 + (size_t)i * C2);
  const float4* bb  = (const float4*)b2;
  const float4* ww  = (const float4*)Wfc;
  float acc = bfc[0];
#pragma unroll 4
  for (int q = 0; q < C2 / 4; ++q) {
    float4 v = row[q], b = bb[q], w = ww[q];
    float e0 = v.x + b.x; e0 = e0 > 0.f ? e0 : (__expf(e0) - 1.0f);
    float e1 = v.y + b.y; e1 = e1 > 0.f ? e1 : (__expf(e1) - 1.0f);
    float e2 = v.z + b.z; e2 = e2 > 0.f ? e2 : (__expf(e2) - 1.0f);
    float e3 = v.w + b.w; e3 = e3 > 0.f ? e3 : (__expf(e3) - 1.0f);
    acc += e0 * w.x + e1 * w.y + e2 * w.z + e3 * w.w;
  }
  y[i] = 1.0f / (1.0f + __expf(-acc));
}

// ---------------------------------------------------------------- launch
static inline long cdivl(long a, long b) { return (a + b - 1) / b; }

extern "C" void kernel_launch(void* const* d_in, const int* in_sizes, int n_in,
                              void* d_out, int out_size, void* d_ws, size_t ws_size,
                              hipStream_t stream) {
  const float* x      = (const float*)d_in[0];
  const int*   ei     = (const int*)d_in[1];      // [2 x E]
  const float* W1     = (const float*)d_in[2];
  const float* a_src1 = (const float*)d_in[3];
  const float* a_dst1 = (const float*)d_in[4];
  const float* b1     = (const float*)d_in[5];
  const float* W2     = (const float*)d_in[6];
  const float* a_src2 = (const float*)d_in[7];
  const float* a_dst2 = (const float*)d_in[8];
  const float* b2     = (const float*)d_in[9];
  const float* Wfc    = (const float*)d_in[10];
  const float* bfc    = (const float*)d_in[11];
  float* y = (float*)d_out;

  const int* srcI = ei;
  const int* dstI = ei + EE;

  // workspace carve-up (256B aligned)
  size_t off = 0;
  auto carve = [&](size_t bytes) -> void* {
    void* p = (char*)d_ws + off;
    off += (bytes + 255) & ~(size_t)255;
    return p;
  };
  _Float16* x16    = (_Float16*)carve((size_t)NN * K1PAD * 2);
  _Float16* W1t16  = (_Float16*)carve((size_t)C1 * K1PAD * 2);
  _Float16* W2t16  = (_Float16*)carve((size_t)C2 * C1 * 2);
  float*    h1     = (float*)carve((size_t)NN * C1 * 4);
  float*    out1   = (float*)carve((size_t)NN * C1 * 4);
  _Float16* h1e16  = (_Float16*)carve((size_t)NN * C1 * 2);
  float*    h2     = (float*)carve((size_t)NN * C2 * 4);
  float*    out2   = (float*)carve((size_t)NN * C2 * 4);
  float*    ssrc1  = (float*)carve((size_t)NN * H1 * 4);
  float*    sdst1  = (float*)carve((size_t)NN * H1 * 4);
  float*    m1     = (float*)carve((size_t)NN * H1 * 4);
  float*    z1     = (float*)carve((size_t)NN * H1 * 4);
  float*    ssrc2  = (float*)carve((size_t)NN * H2 * 4);
  float*    sdst2  = (float*)carve((size_t)NN * H2 * 4);
  float*    m2     = (float*)carve((size_t)NN * H2 * 4);
  float*    z2     = (float*)carve((size_t)NN * H2 * 4);
  float*    lg1    = (float*)carve((size_t)ETOT * H1 * 4);
  float*    lg2    = (float*)carve((size_t)ETOT * H2 * 4);
  (void)ws_size; (void)n_in; (void)in_sizes; (void)out_size;

  const int T = 256;

  // ---- prep: fp16 conversions
  k_convert_x_pad<<<cdivl((long)NN * K1PAD, T), T, 0, stream>>>(x, x16, NN, 27, K1PAD);
  k_convert_wt<<<cdivl((long)C1 * K1PAD, T), T, 0, stream>>>(W1, W1t16, 27, C1, K1PAD);
  k_convert_wt<<<cdivl((long)C2 * C1, T), T, 0, stream>>>(W2, W2t16, C1, C2, C1);

  // ---- layer 1: GEMM (WMMA), scores, softmax-aggregate
  k_gemm_wmma<<<dim3(NN / 16, C1 / 128), T, 0, stream>>>(x16, W1t16, h1, NN, K1PAD, C1);
  k_scores<<<cdivl((long)NN * H1, T), T, 0, stream>>>(h1, a_src1, a_dst1, ssrc1, sdst1, NN, H1);
  k_fill<<<cdivl((long)NN * H1, T), T, 0, stream>>>(m1, -1e30f, (long)NN * H1);
  k_fill<<<cdivl((long)NN * H1, T), T, 0, stream>>>(z1, 0.0f, (long)NN * H1);
  k_fill<<<cdivl((long)NN * C1, T), T, 0, stream>>>(out1, 0.0f, (long)NN * C1);
  k_edge_max<<<cdivl((long)ETOT * H1, T), T, 0, stream>>>(srcI, dstI, ssrc1, sdst1, lg1, m1, H1);
  k_edge_expsum<<<cdivl((long)ETOT * H1, T), T, 0, stream>>>(dstI, lg1, m1, z1, H1);
  k_edge_aggregate<<<ETOT, C1, 0, stream>>>(srcI, dstI, lg1, z1, h1, out1, H1);
  k_elu_bias_f16<<<cdivl((long)NN * C1, T), T, 0, stream>>>(out1, b1, h1e16, NN, C1);

  // ---- layer 2
  k_gemm_wmma<<<dim3(NN / 16, 1), T, 0, stream>>>(h1e16, W2t16, h2, NN, C1, C2);
  k_scores<<<cdivl((long)NN * H2, T), T, 0, stream>>>(h2, a_src2, a_dst2, ssrc2, sdst2, NN, H2);
  k_fill<<<cdivl((long)NN * H2, T), T, 0, stream>>>(m2, -1e30f, (long)NN * H2);
  k_fill<<<cdivl((long)NN * H2, T), T, 0, stream>>>(z2, 0.0f, (long)NN * H2);
  k_fill<<<cdivl((long)NN * C2, T), T, 0, stream>>>(out2, 0.0f, (long)NN * C2);
  k_edge_max<<<cdivl((long)ETOT * H2, T), T, 0, stream>>>(srcI, dstI, ssrc2, sdst2, lg2, m2, H2);
  k_edge_expsum<<<cdivl((long)ETOT * H2, T), T, 0, stream>>>(dstI, lg2, m2, z2, H2);
  k_edge_aggregate<<<ETOT, C2, 0, stream>>>(srcI, dstI, lg2, z2, h2, out2, H2);

  // ---- final fc + sigmoid (bias+elu fused)
  k_final<<<cdivl(NN, T), T, 0, stream>>>(out2, b2, Wfc, bfc, y, NN);
}
